// ExtractLayer_42623255445874
// MI455X (gfx1250) — matrix operand order repaired and checked by reading, not compiled, not run on hardware
//
#include <hip/hip_runtime.h>
#include <math.h>

#define N_OP 100000
#define N_M  20000
#define N_A  20000
#define CDIM 64
#define NEG_SLOPE 0.2f
#define BN_EPS 1e-5f

typedef __attribute__((ext_vector_type(16))) _Float16 v16h;
typedef __attribute__((ext_vector_type(8)))  float    v8f;

// ---------- helpers ----------
__device__ __forceinline__ unsigned f2ord(float f) {
    unsigned u = __float_as_uint(f);
    return (u & 0x80000000u) ? ~u : (u | 0x80000000u);
}
__device__ __forceinline__ float ord2f(unsigned u) {
    return (u & 0x80000000u) ? __uint_as_float(u & 0x7fffffffu) : __uint_as_float(~u);
}
__device__ __forceinline__ float leaky(float v) { return v > 0.f ? v : NEG_SLOPE * v; }

// ---------- WMMA GEMM: Y[N,64] = X[N,64] @ W[64,64] (f16 multiplicands, f32 acc) ----------
__global__ void __launch_bounds__(128)
gemm_xw_wmma(const float* __restrict__ X, const float* __restrict__ W,
             float* __restrict__ Y, int N) {
    const int lane = threadIdx.x & 31;
    const int wave = threadIdx.x >> 5;
    const int r0   = blockIdx.x * 64 + wave * 16;

    const int mrow = lane & 15;
    int row = r0 + mrow; if (row > N - 1) row = N - 1;   // clamp: keep EXEC all-ones
    const float* xrow = X + (size_t)row * CDIM;

    const int koffA = (lane < 16) ? 0 : 8;    // A 16x32 f16 layout
    const int koffB = (lane < 16) ? 0 : 8;    // B 32x16 f16 layout
    const int ncol  = lane & 15;

    v8f c0 = {}, c1 = {}, c2 = {}, c3 = {};

    #pragma unroll
    for (int k0 = 0; k0 < CDIM; k0 += 32) {
        v16h a;
        #pragma unroll
        for (int i = 0; i < 8; ++i) a[i]     = (_Float16)xrow[k0 + koffA + i];
        #pragma unroll
        for (int i = 0; i < 8; ++i) a[8 + i] = (_Float16)xrow[k0 + koffA + 16 + i];

        v16h b0, b1, b2, b3;
        #pragma unroll
        for (int i = 0; i < 16; ++i) {
            const float* wr = W + (size_t)(k0 + koffB + (i & 7) + ((i >> 3) << 4)) * CDIM + ncol;
            b0[i] = (_Float16)wr[0];
            b1[i] = (_Float16)wr[16];
            b2[i] = (_Float16)wr[32];
            b3[i] = (_Float16)wr[48];
        }
        c0 = __builtin_amdgcn_wmma_f32_16x16x32_f16(false, a, false, b0, (short)0, c0, false, false);
        c1 = __builtin_amdgcn_wmma_f32_16x16x32_f16(false, a, false, b1, (short)0, c1, false, false);
        c2 = __builtin_amdgcn_wmma_f32_16x16x32_f16(false, a, false, b2, (short)0, c2, false, false);
        c3 = __builtin_amdgcn_wmma_f32_16x16x32_f16(false, a, false, b3, (short)0, c3, false, false);
    }

    const int mbase = (lane < 16) ? 0 : 8;     // C/D f32 layout
    #pragma unroll
    for (int v = 0; v < 8; ++v) {
        int orow = r0 + mbase + v;
        if (orow < N) {
            float* yr = Y + (size_t)orow * CDIM + ncol;
            yr[0]  = c0[v];
            yr[16] = c1[v];
            yr[32] = c2[v];
            yr[48] = c3[v];
        }
    }
}

// ---------- init accumulator: acc = x + b0 (+ b1) ----------
__global__ void init_acc_kernel(float* __restrict__ acc, const float* __restrict__ x,
                                const float* __restrict__ b0, const float* __restrict__ b1,
                                int N) {
    long i = (long)blockIdx.x * blockDim.x + threadIdx.x;
    long total = (long)N * CDIM;
    if (i >= total) return;
    int c = (int)(i & 63);
    float v = x[i] + b0[c];
    if (b1) v += b1[c];
    acc[i] = v;
}

// ---------- init segment max / denom ----------
__global__ void init_mden_kernel(unsigned* __restrict__ m, float* __restrict__ den, int N) {
    int i = blockIdx.x * blockDim.x + threadIdx.x;
    if (i >= N) return;
    m[i]   = 0x007FFFFFu;   // f2ord(-inf)
    den[i] = 0.f;
}

// ---------- pass A: logits + segment max ----------
__global__ void edge_logits_kernel(const int* __restrict__ src, const int* __restrict__ dst,
                                   const float* __restrict__ xl, const float* __restrict__ xr,
                                   const float* __restrict__ att,
                                   float* __restrict__ e, unsigned* __restrict__ m, int E) {
    int i = blockIdx.x * blockDim.x + threadIdx.x;
    if (i >= E) return;
    int s = src[i], d = dst[i];
    const float4* pl = (const float4*)(xl + (size_t)s * CDIM);
    const float4* pr = (const float4*)(xr + (size_t)d * CDIM);
    const float4* pa = (const float4*)att;
    float acc = 0.f;
    #pragma unroll
    for (int j = 0; j < 16; ++j) {
        float4 l = pl[j], r = pr[j], a = pa[j];
        acc += leaky(l.x + r.x) * a.x + leaky(l.y + r.y) * a.y
             + leaky(l.z + r.z) * a.z + leaky(l.w + r.w) * a.w;
    }
    e[i] = acc;
    atomicMax(&m[d], f2ord(acc));
}

// ---------- pass B: exp(e - m) + segment sum ----------
__global__ void edge_exp_kernel(const int* __restrict__ dst, float* __restrict__ e,
                                const unsigned* __restrict__ m, float* __restrict__ den, int E) {
    int i = blockIdx.x * blockDim.x + threadIdx.x;
    if (i >= E) return;
    int d = dst[i];
    float mv = ord2f(m[d]);
    if (!(mv > -INFINITY)) mv = 0.f;     // empty-segment convention (matches reference)
    float ex = __expf(e[i] - mv);
    e[i] = ex;
    atomicAdd(&den[d], ex);
}

// ---------- pass C: acc[dst] += alpha * xl[src] (4 threads / edge, 16 ch each) ----------
__global__ void edge_scatter_kernel(const int* __restrict__ src, const int* __restrict__ dst,
                                    const float* __restrict__ xl, const float* __restrict__ e,
                                    const float* __restrict__ den, float* __restrict__ acc, int E) {
    int t = blockIdx.x * blockDim.x + threadIdx.x;
    int i = t >> 2;
    if (i >= E) return;
    int q = (t & 3) * 16;
    int s = src[i], d = dst[i];
    float alpha = e[i] / fmaxf(den[d], 1e-16f);
    const float* xls = xl + (size_t)s * CDIM + q;
    float*       ad  = acc + (size_t)d * CDIM + q;
    #pragma unroll
    for (int c = 0; c < 16; ++c) atomicAdd(&ad[c], alpha * xls[c]);
}

// ---------- BN stats: per-channel sum / sumsq ----------
__global__ void zero_stats_kernel(float* __restrict__ stats, int n) {
    int i = blockIdx.x * blockDim.x + threadIdx.x;
    if (i < n) stats[i] = 0.f;
}

__global__ void __launch_bounds__(256)
bn_stats_kernel(const float* __restrict__ v, int N, float* __restrict__ stats) {
    __shared__ float ssum[256];
    __shared__ float ssq[256];
    long total  = (long)N * CDIM;
    long stride = (long)gridDim.x * blockDim.x;   // multiple of 64 -> channel fixed per thread
    float s = 0.f, s2 = 0.f;
    for (long i = (long)blockIdx.x * blockDim.x + threadIdx.x; i < total; i += stride) {
        float x = v[i];
        s += x; s2 += x * x;
    }
    ssum[threadIdx.x] = s;
    ssq[threadIdx.x]  = s2;
    __syncthreads();
    if (threadIdx.x < 64) {
        float a = ssum[threadIdx.x] + ssum[threadIdx.x + 64] + ssum[threadIdx.x + 128] + ssum[threadIdx.x + 192];
        float b = ssq[threadIdx.x]  + ssq[threadIdx.x + 64]  + ssq[threadIdx.x + 128]  + ssq[threadIdx.x + 192];
        atomicAdd(&stats[threadIdx.x],      a);
        atomicAdd(&stats[64 + threadIdx.x], b);
    }
}

// ---------- BN normalize (in place) ----------
__global__ void bn_norm_kernel(float* __restrict__ v, int N, const float* __restrict__ stats,
                               const float* __restrict__ gamma, const float* __restrict__ beta) {
    long i = (long)blockIdx.x * blockDim.x + threadIdx.x;
    long total = (long)N * CDIM;
    if (i >= total) return;
    int c = (int)(i & 63);
    float inv_n = 1.f / (float)N;
    float mu  = stats[c] * inv_n;
    float var = stats[64 + c] * inv_n - mu * mu;
    float x = v[i];
    v[i] = gamma[c] * (x - mu) * rsqrtf(var + BN_EPS) + beta[c];
}

// ---------- host orchestration ----------
extern "C" void kernel_launch(void* const* d_in, const int* in_sizes, int n_in,
                              void* d_out, int out_size, void* d_ws, size_t ws_size,
                              hipStream_t stream) {
    const float* x_op  = (const float*)d_in[0];
    const float* x_mch = (const float*)d_in[1];
    const float* x_agv = (const float*)d_in[2];
    const int* ei_pred = (const int*)d_in[3];
    const int* ei_succ = (const int*)d_in[4];
    const int* ei_om   = (const int*)d_in[5];
    const int* ei_oa   = (const int*)d_in[6];
    const int* ei_ma   = (const int*)d_in[7];
    const float* Wl    = (const float*)d_in[8];
    const float* Wr    = (const float*)d_in[9];
    const float* att   = (const float*)d_in[10];
    const float* bias  = (const float*)d_in[11];
    const float* bn_g  = (const float*)d_in[12];
    const float* bn_b  = (const float*)d_in[13];

    const int E_pred = in_sizes[3] / 2;
    const int E_succ = in_sizes[4] / 2;
    const int E_om   = in_sizes[5] / 2;
    const int E_oa   = in_sizes[6] / 2;
    const int E_ma   = in_sizes[7] / 2;

    // workspace layout (floats)
    float* ws      = (float*)d_ws;
    float* xlbuf   = ws;                                 // N_OP*64
    float* xrbuf   = xlbuf + (size_t)N_OP * CDIM;        // N_OP*64
    float* ebuf    = xrbuf + (size_t)N_OP * CDIM;        // max E (800000)
    unsigned* mbuf = (unsigned*)(ebuf + 800000);         // N_OP
    float* denbuf  = (float*)(mbuf + N_OP);              // N_OP
    float* stats   = denbuf + N_OP;                      // 3 * 128

    float* out_op = (float*)d_out;
    float* out_m  = out_op + (size_t)N_OP * CDIM;
    float* out_a  = out_m  + (size_t)N_M  * CDIM;

    const int TB = 256;
    auto eceil = [](long n, int b) { return (unsigned)((n + b - 1) / b); };

    // accumulators = residual + bias (HeteroConv aggr='sum' biases fold in)
    init_acc_kernel<<<eceil((long)N_OP * CDIM, TB), TB, 0, stream>>>(out_op, x_op,  bias + 0 * CDIM, bias + 1 * CDIM, N_OP);
    init_acc_kernel<<<eceil((long)N_M  * CDIM, TB), TB, 0, stream>>>(out_m,  x_mch, bias + 2 * CDIM, nullptr,         N_M);
    init_acc_kernel<<<eceil((long)N_A  * CDIM, TB), TB, 0, stream>>>(out_a,  x_agv, bias + 3 * CDIM, bias + 4 * CDIM, N_A);

    struct ET {
        const float* xs; int ns;
        const float* xd; int nd;
        const int* src; const int* dst; int e;
        int w; float* acc;
    };
    const ET ets[5] = {
        { x_op,  N_OP, x_op,  N_OP, ei_pred, ei_pred + E_pred, E_pred, 0, out_op },
        { x_op,  N_OP, x_op,  N_OP, ei_succ, ei_succ + E_succ, E_succ, 1, out_op },
        { x_op,  N_OP, x_mch, N_M,  ei_om,   ei_om   + E_om,   E_om,   2, out_m  },
        { x_op,  N_OP, x_agv, N_A,  ei_oa,   ei_oa   + E_oa,   E_oa,   3, out_a  },
        { x_mch, N_M,  x_agv, N_A,  ei_ma,   ei_ma   + E_ma,   E_ma,   4, out_a  },
    };

    for (int t = 0; t < 5; ++t) {
        const ET& et = ets[t];
        // dense transforms via WMMA
        gemm_xw_wmma<<<(et.ns + 63) / 64, 128, 0, stream>>>(et.xs, Wl + (size_t)et.w * CDIM * CDIM, xlbuf, et.ns);
        gemm_xw_wmma<<<(et.nd + 63) / 64, 128, 0, stream>>>(et.xd, Wr + (size_t)et.w * CDIM * CDIM, xrbuf, et.nd);
        // segment softmax (3 passes)
        init_mden_kernel<<<eceil(et.nd, TB), TB, 0, stream>>>(mbuf, denbuf, et.nd);
        edge_logits_kernel<<<eceil(et.e, TB), TB, 0, stream>>>(et.src, et.dst, xlbuf, xrbuf,
                                                               att + (size_t)et.w * CDIM, ebuf, mbuf, et.e);
        edge_exp_kernel<<<eceil(et.e, TB), TB, 0, stream>>>(et.dst, ebuf, mbuf, denbuf, et.e);
        edge_scatter_kernel<<<eceil((long)et.e * 4, TB), TB, 0, stream>>>(et.src, et.dst, xlbuf,
                                                                          ebuf, denbuf, et.acc, et.e);
    }

    // per-type BatchNorm1d (training mode), in place on d_out
    zero_stats_kernel<<<2, TB, 0, stream>>>(stats, 3 * 128);
    bn_stats_kernel<<<256, 256, 0, stream>>>(out_op, N_OP, stats + 0 * 128);
    bn_stats_kernel<<<128, 256, 0, stream>>>(out_m,  N_M,  stats + 1 * 128);
    bn_stats_kernel<<<128, 256, 0, stream>>>(out_a,  N_A,  stats + 2 * 128);
    bn_norm_kernel<<<eceil((long)N_OP * CDIM, TB), TB, 0, stream>>>(out_op, N_OP, stats + 0 * 128, bn_g + 0 * CDIM, bn_b + 0 * CDIM);
    bn_norm_kernel<<<eceil((long)N_M  * CDIM, TB), TB, 0, stream>>>(out_m,  N_M,  stats + 1 * 128, bn_g + 1 * CDIM, bn_b + 1 * CDIM);
    bn_norm_kernel<<<eceil((long)N_A  * CDIM, TB), TB, 0, stream>>>(out_a,  N_A,  stats + 2 * 128, bn_g + 2 * CDIM, bn_b + 2 * CDIM);
}
